// GCN_1760936591397
// MI455X (gfx1250) — compile-verified, matrix operand order checked by reading, not run on hardware
//
#include <hip/hip_runtime.h>
#include <cstdint>
#include <cstddef>

// ---------------------------------------------------------------------------
// CDNA5 (gfx1250) GCN forward: WMMA bf16 GEMMs (LDS-staged B panel, ping-pong
// A buffers) + atomic scatter aggregation, BN, log-softmax.
// ---------------------------------------------------------------------------

typedef __bf16 bf16_t;
typedef bf16_t v16bf __attribute__((ext_vector_type(16)));
typedef float  v8f   __attribute__((ext_vector_type(8)));

#define BN_EPS 1e-5f

__device__ __forceinline__ unsigned short f2bf(float f) {
  // round-to-nearest-even f32 -> bf16
  unsigned int u = __float_as_uint(f);
  u += 0x7FFFu + ((u >> 16) & 1u);
  return (unsigned short)(u >> 16);
}

// ---------------- degree / normalization ----------------
__global__ void k_deg(const int* __restrict__ dst, float* __restrict__ deg, int E) {
  int e = blockIdx.x * blockDim.x + threadIdx.x;
  if (e < E) atomicAdd(&deg[dst[e]], 1.0f);
}

__global__ void k_dinv(float* __restrict__ deg, int N) {
  int i = blockIdx.x * blockDim.x + threadIdx.x;
  if (i < N) deg[i] = rsqrtf(deg[i] + 1.0f);   // self-loop included; deg>0 always
}

// ---------------- fragment packing (f32 -> bf16, WMMA layout) ----------------
// A fragment layout (16-bit A 16x32, ISA 7.12.2):
//   lane<16 : M=lane, halves j=0..7 -> K=half*8+j, j=8..15 -> K=16+half*8+(j-8)
// packed as [ (mt*ktiles + kt)*32 + lane ][16 halves] contiguous.
__global__ void k_pack_a(const float* __restrict__ X, unsigned short* __restrict__ A,
                         int Nrows, int K, long long total) {
  long long idx = (long long)blockIdx.x * blockDim.x + threadIdx.x;
  if (idx >= total) return;
  int j          = (int)(idx & 15);
  long long frag = idx >> 4;
  int lane       = (int)(frag & 31);
  long long tile = frag >> 5;
  int ktiles     = K >> 5;
  int kt         = (int)(tile % ktiles);
  long long mt   = tile / ktiles;
  int row        = (int)(mt * 16 + (lane & 15));
  int half       = lane >> 4;
  int k          = kt * 32 + ((j < 8) ? (half * 8 + j) : (16 + half * 8 + (j - 8)));
  float v        = (row < Nrows) ? X[(size_t)row * K + k] : 0.0f;
  A[idx] = f2bf(v);
}

// B fragment: lane<16 : N=lane, same K-half striping as A.
// packed as [ (kt*nTiles + nt)*32 + lane ][16 halves].
__global__ void k_pack_w(const float* __restrict__ W, unsigned short* __restrict__ O,
                         int K, int NoutReal, int NoutPad, long long total) {
  long long idx = (long long)blockIdx.x * blockDim.x + threadIdx.x;
  if (idx >= total) return;
  int j          = (int)(idx & 15);
  long long frag = idx >> 4;
  int lane       = (int)(frag & 31);
  long long tile = frag >> 5;
  int nTiles     = NoutPad >> 4;
  int nt         = (int)(tile % nTiles);
  int kt         = (int)(tile / nTiles);
  int n          = nt * 16 + (lane & 15);
  int half       = lane >> 4;
  int k          = kt * 32 + ((j < 8) ? (half * 8 + j) : (16 + half * 8 + (j - 8)));
  float v        = (n < NoutReal) ? W[(size_t)k * NoutReal + n] : 0.0f;
  O[idx] = f2bf(v);
}

// ---------------- WMMA GEMM: C[Mpad x NoutPad] = A(bf16) @ W(bf16) ----------------
// Block = 8 waves, one 64-column group. B panel (<=32KB) staged in LDS once and
// reused by 8 waves x T M-tiles each. A k-tiles ping-pong double-buffered from
// global (two buffers loaded 2 k-steps ahead -> no register rotation copies).
template<int NW, int T>
__global__ void k_gemm_wmma(const unsigned short* __restrict__ Apack,
                            const unsigned short* __restrict__ Wp,
                            float* __restrict__ C,
                            int ktiles, int nTilesTot, int mTiles) {
  __shared__ __align__(16) unsigned short Bs[8 * NW * 32 * 16];  // max ktiles = 8
  const int tid    = threadIdx.x;
  const int lane   = tid & 31;
  const int waveId = tid >> 5;
  const int ng     = blockIdx.y;

  // ---- cooperative B-panel copy: 4KB contiguous per k-tile ----
  {
    const uint4* src = reinterpret_cast<const uint4*>(Wp);
    uint4*       dst = reinterpret_cast<uint4*>(Bs);
    const int vecsPerKt = NW * 32 * 16 / 8;                     // 256 uint4 / ktile
    for (int i = tid; i < ktiles * vecsPerKt; i += (int)blockDim.x) {
      int kt = i / vecsPerKt;
      int o  = i - kt * vecsPerKt;
      dst[(size_t)kt * vecsPerKt + o] =
          src[((size_t)kt * nTilesTot + (size_t)ng * NW) * 64 + o];
    }
  }
  __syncthreads();

  const v16bf* Bsf   = reinterpret_cast<const v16bf*>(Bs);
  const v16bf* Abase = reinterpret_cast<const v16bf*>(Apack);

  v8f zero = {0.f, 0.f, 0.f, 0.f, 0.f, 0.f, 0.f, 0.f};
  const int ldc = nTilesTot * 16;

  for (int t = 0; t < T; ++t) {
    int mt = (blockIdx.x * (int)(blockDim.x >> 5) + waveId) * T + t;
    if (mt >= mTiles) continue;                 // wave-uniform: EXEC stays all-ones

    const v16bf* Af = Abase + (size_t)mt * ktiles * 32 + lane;
    if (mt + 1 < mTiles)
      __builtin_prefetch((const void*)(Af + (size_t)ktiles * 32), 0, 3);

    v8f acc[NW];
#pragma unroll
    for (int w = 0; w < NW; ++w) acc[w] = zero;

    // ktiles is always even (4 or 8): ping-pong A fragments a0/a1.
    v16bf a0 = Af[0];
    v16bf a1 = Af[32];
    for (int kt = 0; kt < ktiles; kt += 2) {
      const v16bf* b0 = Bsf + (size_t)kt * (NW * 32) + lane;
#pragma unroll
      for (int w = 0; w < NW; ++w) {
        acc[w] = __builtin_amdgcn_wmma_f32_16x16x32_bf16(
            false, a0, false, b0[(size_t)w * 32], (short)0, acc[w], false, false);
      }
      if (kt + 2 < ktiles) a0 = Af[(size_t)(kt + 2) * 32];

      const v16bf* b1 = b0 + NW * 32;
#pragma unroll
      for (int w = 0; w < NW; ++w) {
        acc[w] = __builtin_amdgcn_wmma_f32_16x16x32_bf16(
            false, a1, false, b1[(size_t)w * 32], (short)0, acc[w], false, false);
      }
      if (kt + 3 < ktiles) a1 = Af[(size_t)(kt + 3) * 32];
    }

    // C/D layout: VGPR r -> M = r (lanes 0-15) / r+8 (lanes 16-31); N = lane&15.
    int m0 = mt * 16 + ((lane >> 4) << 3);
    int nc = ng * (NW * 16) + (lane & 15);
#pragma unroll
    for (int w = 0; w < NW; ++w) {
#pragma unroll
      for (int r = 0; r < 8; ++r) {
        C[(size_t)(m0 + r) * ldc + nc + w * 16] = acc[w][r];
      }
    }
  }
}

// ---------------- edge scatter: Ha[dst] += Hg[src] * dinv[src]*dinv[dst] ----------------
__global__ void k_scatter(const int* __restrict__ ei, const float* __restrict__ Hg,
                          const float* __restrict__ dinv, float* __restrict__ Ha,
                          int E, int Fpad, int chunks, long long total) {
  long long t = (long long)blockIdx.x * blockDim.x + threadIdx.x;
  if (t >= total) return;
  int e = (int)(t / chunks);
  int c = (int)(t - (long long)e * chunks) * 4;
  int s = ei[e];
  int d = ei[E + e];
  float w = dinv[s] * dinv[d];
  const float4 v = *reinterpret_cast<const float4*>(Hg + (size_t)s * Fpad + c);
  float* o = Ha + (size_t)d * Fpad + c;
  atomicAdd(o + 0, v.x * w);
  atomicAdd(o + 1, v.y * w);
  atomicAdd(o + 2, v.z * w);
  atomicAdd(o + 3, v.w * w);
}

// self-loop contribution + bias
__global__ void k_selfbias(const float* __restrict__ Hg, const float* __restrict__ dinv,
                           const float* __restrict__ b, float* __restrict__ Ha,
                           int Fpad, int Fv, long long total) {
  long long idx = (long long)blockIdx.x * blockDim.x + threadIdx.x;
  if (idx >= total) return;
  int row = (int)(idx / Fv);
  int c   = (int)(idx - (long long)row * Fv);
  float di = dinv[row];
  size_t o = (size_t)row * Fpad + c;
  Ha[o] += Hg[o] * di * di + b[c];
}

// ---------------- batchnorm ----------------
__global__ void k_bnstats(const float* __restrict__ H, float* __restrict__ stats,
                          int N, int F, int rowsPerBlock) {
  int c  = threadIdx.x;                       // blockDim.x == F (256)
  int r0 = blockIdx.x * rowsPerBlock;
  int r1 = r0 + rowsPerBlock; if (r1 > N) r1 = N;
  float s = 0.f, s2 = 0.f;
  for (int r = r0; r < r1; ++r) {
    float v = H[(size_t)r * F + c];
    s += v; s2 += v * v;
  }
  atomicAdd(&stats[c], s);
  atomicAdd(&stats[F + c], s2);
}

__global__ void k_bnapply(const float* __restrict__ H, const float* __restrict__ stats,
                          const float* __restrict__ g, const float* __restrict__ be,
                          const float* __restrict__ res, float* __restrict__ out,
                          float invN, int F, long long total) {
  long long idx = (long long)blockIdx.x * blockDim.x + threadIdx.x;
  if (idx >= total) return;
  int row = (int)(idx / F);
  int c   = (int)(idx - (long long)row * F);
  float mean = stats[c] * invN;
  float var  = stats[F + c] * invN - mean * mean;
  size_t o = (size_t)row * F + c;
  float v = (H[o] - mean) * rsqrtf(var + BN_EPS) * g[c] + be[c];
  v = fmaxf(v, 0.f);
  if (res) v += res[o];
  out[o] = v;
}

// ---------------- log-softmax over C classes (padded stride) ----------------
__global__ void k_logsoftmax(const float* __restrict__ in, float* __restrict__ out,
                             int N, int Fpad, int C) {
  int r = blockIdx.x * blockDim.x + threadIdx.x;
  if (r >= N) return;
  const float* p = in + (size_t)r * Fpad;
  float mx = p[0];
  for (int c = 1; c < C; ++c) mx = fmaxf(mx, p[c]);
  float s = 0.f;
  for (int c = 0; c < C; ++c) s += expf(p[c] - mx);
  float ls = logf(s);
  for (int c = 0; c < C; ++c) out[(size_t)r * C + c] = p[c] - mx - ls;
}

// ---------------------------------------------------------------------------

static inline unsigned gblocks(long long total, int tpb) {
  return (unsigned)((total + tpb - 1) / tpb);
}

extern "C" void kernel_launch(void* const* d_in, const int* in_sizes, int n_in,
                              void* d_out, int out_size, void* d_ws, size_t ws_size,
                              hipStream_t stream) {
  const float* x   = (const float*)d_in[0];
  const int*   ei  = (const int*)d_in[1];
  const float* W1  = (const float*)d_in[2];
  const float* b1  = (const float*)d_in[3];
  const float* g1  = (const float*)d_in[4];
  const float* be1 = (const float*)d_in[5];
  const float* W2  = (const float*)d_in[6];
  const float* b2  = (const float*)d_in[7];
  const float* g2  = (const float*)d_in[8];
  const float* be2 = (const float*)d_in[9];
  const float* W3  = (const float*)d_in[10];
  const float* b3  = (const float*)d_in[11];
  float* out = (float*)d_out;

  const int N = in_sizes[0] / 128;      // 169343
  const int E = in_sizes[1] / 2;        // 1200000
  const size_t Mpad = ((size_t)N + 15) & ~(size_t)15;
  const int mTiles = (int)(Mpad / 16);
  const int TPB = 256;

  // workspace carve
  char* p = (char*)d_ws;
  auto carve = [&](size_t bytes) -> char* {
    char* r = p; p += (bytes + 255) & ~(size_t)255; return r;
  };
  float*          dinv  = (float*)carve(Mpad * 4);
  float*          stats = (float*)carve(512 * 4);
  unsigned short* Apack = (unsigned short*)carve(Mpad * 256 * 2);
  unsigned short* Wp    = (unsigned short*)carve(256 * 256 * 2);
  float*          Hg    = (float*)carve(Mpad * 256 * 4);   // GEMM result
  float*          Ha    = (float*)carve(Mpad * 256 * 4);   // aggregated
  float*          Hp    = (float*)carve(Mpad * 256 * 4);   // activation / residual

  // ---- degrees: deg[dst]+=1 over edges, then dinv = rsqrt(deg+1) ----
  hipMemsetAsync(dinv, 0, Mpad * 4, stream);
  k_deg<<<gblocks(E, TPB), TPB, 0, stream>>>(ei + E, dinv, E);
  k_dinv<<<gblocks(N, TPB), TPB, 0, stream>>>(dinv, N);

  const int T = 4;  // M tiles per wave in GEMM
  const unsigned gx = gblocks(mTiles, 8 * T);

  // ================= Layer 1: x(128) -> 256, BN, ReLU =================
  {
    const int K = 128, Fpad = 256, Fv = 256;
    long long totA = (long long)Mpad * K;
    k_pack_a<<<gblocks(totA, TPB), TPB, 0, stream>>>(x, Apack, N, K, totA);
    long long totW = (long long)K * Fpad;
    k_pack_w<<<gblocks(totW, TPB), TPB, 0, stream>>>(W1, Wp, K, 256, Fpad, totW);
    int ktiles = K / 32, nT = Fpad / 16;
    dim3 grid(gx, nT / 4);
    k_gemm_wmma<4, 4><<<grid, 256, 0, stream>>>(Apack, Wp, Hg, ktiles, nT, mTiles);
    hipMemsetAsync(Ha, 0, Mpad * Fpad * 4, stream);
    long long totS = (long long)E * (Fv / 4);
    k_scatter<<<gblocks(totS, TPB), TPB, 0, stream>>>(ei, Hg, dinv, Ha, E, Fpad, Fv / 4, totS);
    long long totB = (long long)N * Fv;
    k_selfbias<<<gblocks(totB, TPB), TPB, 0, stream>>>(Hg, dinv, b1, Ha, Fpad, Fv, totB);
    hipMemsetAsync(stats, 0, 512 * 4, stream);
    k_bnstats<<<gblocks(N, 512), 256, 0, stream>>>(Ha, stats, N, 256, 512);
    k_bnapply<<<gblocks(totB, TPB), TPB, 0, stream>>>(Ha, stats, g1, be1, nullptr, Hp,
                                                      1.0f / (float)N, 256, totB);
  }

  // ================= Layer 2: 256 -> 256, BN, ReLU, +residual =================
  {
    const int K = 256, Fpad = 256, Fv = 256;
    long long totA = (long long)Mpad * K;
    k_pack_a<<<gblocks(totA, TPB), TPB, 0, stream>>>(Hp, Apack, N, K, totA);
    long long totW = (long long)K * Fpad;
    k_pack_w<<<gblocks(totW, TPB), TPB, 0, stream>>>(W2, Wp, K, 256, Fpad, totW);
    int ktiles = K / 32, nT = Fpad / 16;
    dim3 grid(gx, nT / 4);
    k_gemm_wmma<4, 4><<<grid, 256, 0, stream>>>(Apack, Wp, Hg, ktiles, nT, mTiles);
    hipMemsetAsync(Ha, 0, Mpad * Fpad * 4, stream);
    long long totS = (long long)E * (Fv / 4);
    k_scatter<<<gblocks(totS, TPB), TPB, 0, stream>>>(ei, Hg, dinv, Ha, E, Fpad, Fv / 4, totS);
    long long totB = (long long)N * Fv;
    k_selfbias<<<gblocks(totB, TPB), TPB, 0, stream>>>(Hg, dinv, b2, Ha, Fpad, Fv, totB);
    hipMemsetAsync(stats, 0, 512 * 4, stream);
    k_bnstats<<<gblocks(N, 512), 256, 0, stream>>>(Ha, stats, N, 256, 512);
    // h2 = relu(bn(Ha)) + h1(Hp)  -> store into Hg
    k_bnapply<<<gblocks(totB, TPB), TPB, 0, stream>>>(Ha, stats, g2, be2, Hp, Hg,
                                                      1.0f / (float)N, 256, totB);
  }

  // ================= Layer 3: 256 -> 40 (padded 64), log_softmax =================
  {
    const int K = 256, Fpad = 64, Fv = 40;
    long long totA = (long long)Mpad * K;
    k_pack_a<<<gblocks(totA, TPB), TPB, 0, stream>>>(Hg, Apack, N, K, totA);
    long long totW = (long long)K * Fpad;
    k_pack_w<<<gblocks(totW, TPB), TPB, 0, stream>>>(W3, Wp, K, 40, Fpad, totW);
    int ktiles = K / 32, nT = Fpad / 16;
    dim3 grid(gx, 1);   // 64 cols -> single column group
    k_gemm_wmma<4, 4><<<grid, 256, 0, stream>>>(Apack, Wp, Ha, ktiles, nT, mTiles);
    hipMemsetAsync(Hp, 0, Mpad * Fpad * 4, stream);
    long long totS = (long long)E * (Fv / 4);
    k_scatter<<<gblocks(totS, TPB), TPB, 0, stream>>>(ei, Ha, dinv, Hp, E, Fpad, Fv / 4, totS);
    long long totB = (long long)N * Fv;
    k_selfbias<<<gblocks(totB, TPB), TPB, 0, stream>>>(Ha, dinv, b3, Hp, Fpad, Fv, totB);
    k_logsoftmax<<<gblocks(N, TPB), TPB, 0, stream>>>(Hp, out, N, Fpad, Fv);
  }
}
